// PointNet_SA_Layer_4114578669988
// MI455X (gfx1250) — matrix operand order repaired and checked by reading, not compiled, not run on hardware
//
#include <hip/hip_runtime.h>
#include <hip/hip_bf16.h>

// ---------------- problem constants (match reference) ----------------
#define BB      4
#define NN      8192
#define DD      64
#define SS      2048
#define KK      32
#define NCOL    ((long)BB * SS * KK)      // 262144
#define CIN1P   96                        // 67 padded to 3*32
#define CIN2P   64
#define COUT12  64
#define COUTL   128
#define RADIUS_ 0.2f
#define EPS_    1e-5f

typedef _Float16 v16h __attribute__((ext_vector_type(16)));
typedef float    v8f  __attribute__((ext_vector_type(8)));

// ---------------- WMMA fragment loaders (CDNA5 16x16x32 f16) ----------------
// A (16x32, MxK), W stored row-major [Cout][CP]:
//   lane L: M = L&15, kb = (L>=16)?16:0 ; half e -> K = kb+e  (16 contiguous halves)
__device__ inline v16h load_A_frag(const _Float16* __restrict__ Wp, int CP,
                                   int m0, int c0, int lane) {
    int M  = lane & 15;
    int kb = (lane >> 4) * 16;
    const _Float16* p = Wp + (long)(m0 + M) * CP + c0 + kb;
    v16h a;
#pragma unroll
    for (int i = 0; i < 16; ++i) a[i] = p[i];
    return a;
}

// B (32x16, KxN), X stored [n][CP] (channels contiguous):
//   lane L: N = L&15, koff = (L>=16)?16:0 ; half e -> K = koff+e  (16 contiguous halves)
__device__ inline v16h load_B_frag(const _Float16* __restrict__ X, int CP,
                                   long n0, int c0, int lane) {
    int N    = lane & 15;
    int koff = (lane >> 4) * 16;
    const _Float16* p = X + (long)(n0 + N) * CP + c0 + koff;
    v16h b;
#pragma unroll
    for (int i = 0; i < 16; ++i) b[i] = p[i];
    return b;
}

// ---------------- 0) prep: pack weights to padded f16, fold BN ----------------
__global__ void prep_kernel(const float* __restrict__ W1, const float* __restrict__ b1,
                            const float* __restrict__ g1, const float* __restrict__ be1,
                            const float* __restrict__ m1, const float* __restrict__ v1,
                            const float* __restrict__ W2, const float* __restrict__ b2,
                            const float* __restrict__ g2, const float* __restrict__ be2,
                            const float* __restrict__ m2, const float* __restrict__ v2,
                            const float* __restrict__ Wl,
                            _Float16* __restrict__ W1p, _Float16* __restrict__ W2p,
                            _Float16* __restrict__ Wlp,
                            float* __restrict__ bn1a, float* __restrict__ bn1c,
                            float* __restrict__ bn2a, float* __restrict__ bn2c) {
    int t = threadIdx.x;
    for (int i = t; i < COUT12 * CIN1P; i += blockDim.x) {
        int o = i / CIN1P, c = i % CIN1P;
        W1p[i] = (_Float16)(c < 67 ? W1[o * 67 + c] : 0.f);
    }
    for (int i = t; i < COUT12 * CIN2P; i += blockDim.x) W2p[i] = (_Float16)W2[i];
    for (int i = t; i < COUTL * CIN2P; i += blockDim.x)  Wlp[i] = (_Float16)Wl[i];
    if (t < COUT12) {
        float s1 = g1[t] * rsqrtf(v1[t] + EPS_);
        bn1a[t] = s1; bn1c[t] = (b1[t] - m1[t]) * s1 + be1[t];
        float s2 = g2[t] * rsqrtf(v2[t] + EPS_);
        bn2a[t] = s2; bn2c[t] = (b2[t] - m2[t]) * s2 + be2[t];
    }
}

// ---------------- 1) farthest point sampling (one block per batch) ----------------
// Coords + running min-dist in LDS (128KB < 320KB/WGP). Per iteration:
// wave-local argmax via shfl butterfly (no barriers), then a single 8-entry
// LDS handoff -> only 2 __syncthreads per iteration on the serial chain.
__global__ void fps_kernel(const float* __restrict__ xyz, int* __restrict__ fidx,
                           float* __restrict__ sampled) {
    extern __shared__ float smem[];
    float* mind = smem;                 // NN
    float* sx   = mind + NN;            // NN
    float* sy   = sx + NN;              // NN
    float* sz   = sy + NN;              // NN
    float* wval = sz + NN;              // 8 (waves per block)
    int*   widx = (int*)(wval + 8);     // 8
    __shared__ int cur;

    int b    = blockIdx.x;
    int tid  = threadIdx.x;
    int bd   = blockDim.x;
    int lane = tid & 31;
    int wave = tid >> 5;
    const float* X = xyz + (long)b * 3 * NN;

    for (int j = tid; j < NN; j += bd) {
        sx[j] = X[j]; sy[j] = X[NN + j]; sz[j] = X[2 * NN + j];
        mind[j] = 1e10f;
    }
    if (tid == 0) cur = 0;
    __syncthreads();

    for (int i = 0; i < SS; ++i) {
        int c = cur;
        float px = sx[c], py = sy[c], pz = sz[c];
        if (tid == 0) {
            fidx[b * SS + i] = c;
            long o = ((long)b * SS + i) * 3;
            sampled[o + 0] = px; sampled[o + 1] = py; sampled[o + 2] = pz;
        }
        float best = -1.f; int bi = 0x7fffffff;
        for (int j = tid; j < NN; j += bd) {
            float dx = sx[j] - px, dy = sy[j] - py, dz = sz[j] - pz;
            float d  = dx * dx + dy * dy + dz * dz;
            float md = mind[j]; md = md < d ? md : d; mind[j] = md;
            if (md > best) { best = md; bi = j; }
        }
        // intra-wave argmax (lowest index wins ties), no barriers
#pragma unroll
        for (int off = 16; off >= 1; off >>= 1) {
            float ov = __shfl_xor(best, off, 32);
            int   oi = __shfl_xor(bi,   off, 32);
            if (ov > best || (ov == best && oi < bi)) { best = ov; bi = oi; }
        }
        if (lane == 0) { wval[wave] = best; widx[wave] = bi; }
        __syncthreads();
        if (tid == 0) {
            float bv = wval[0]; int bj = widx[0];
#pragma unroll
            for (int w = 1; w < 8; ++w) {
                float ov = wval[w]; int oi = widx[w];
                if (ov > bv || (ov == bv && oi < bj)) { bv = ov; bj = oi; }
            }
            cur = bj;
        }
        __syncthreads();
    }
}

// ---------------- 2) sampled_xyz output transpose (B,S,3)->(B,3,S) ----------------
__global__ void sampled_out_kernel(const float* __restrict__ sampled, float* __restrict__ out) {
    long t = (long)blockIdx.x * blockDim.x + threadIdx.x;
    if (t >= (long)BB * 3 * SS) return;
    int b = (int)(t / (3 * SS));
    int r = (int)(t % (3 * SS));
    int d = r / SS, s = r % SS;
    out[t] = sampled[((long)b * SS + s) * 3 + d];
}

// ---------------- 3) ball query (one thread per centroid) ----------------
__global__ void ball_kernel(const float* __restrict__ xyz, const float* __restrict__ sampled,
                            int* __restrict__ bidx) {
    long t = (long)blockIdx.x * blockDim.x + threadIdx.x;
    if (t >= (long)BB * SS) return;
    int b = (int)(t / SS);
    const float* X = xyz + (long)b * 3 * NN;
    float cx = sampled[t * 3 + 0], cy = sampled[t * 3 + 1], cz = sampled[t * 3 + 2];
    const float r2 = RADIUS_ * RADIUS_;
    int cnt = 0, first = 0;
    int* dst = bidx + t * KK;
    for (int j = 0; j < NN; ++j) {
        float dx = X[j] - cx, dy = X[NN + j] - cy, dz = X[2 * NN + j] - cz;
        if (dx * dx + dy * dy + dz * dz <= r2) {
            if (cnt == 0) first = j;
            dst[cnt++] = j;
            if (cnt >= KK) break;
        }
    }
    for (int k = cnt; k < KK; ++k) dst[k] = first;
}

// ---------------- 4) gather + group -> Xg f16 [n][96] ----------------
__global__ void gather_kernel(const float* __restrict__ xyz, const float* __restrict__ points,
                              const float* __restrict__ sampled, const int* __restrict__ bidx,
                              _Float16* __restrict__ Xg) {
    long t = (long)blockIdx.x * blockDim.x + threadIdx.x;
    if (t >= NCOL) return;
    long bs = t >> 5;                 // (b,s)
    int  b  = (int)(bs / SS);
    int  pid = bidx[t];
    const float* P = points + (long)b * DD * NN;
    _Float16* dst = Xg + t * CIN1P;
#pragma unroll 8
    for (int c = 0; c < DD; ++c) dst[c] = (_Float16)P[(long)c * NN + pid];
    const float* X   = xyz + (long)b * 3 * NN;
    const float* ctr = sampled + bs * 3;
    const float inv_r = 1.0f / RADIUS_;
    dst[64] = (_Float16)((X[pid]            - ctr[0]) * inv_r);
    dst[65] = (_Float16)((X[NN + pid]       - ctr[1]) * inv_r);
    dst[66] = (_Float16)((X[2 * NN + pid]   - ctr[2]) * inv_r);
#pragma unroll
    for (int c = 67; c < CIN1P; ++c) dst[c] = (_Float16)0.f;
}

// ---------------- 5) GEMM + BN + ReLU, wave computes ALL 64 out-channels ----------------
// B fragments are loaded ONCE per wave and reused across the 4 output-channel
// tiles (A-matrix sweeps) -> conv1: 12 WMMA/wave, conv2: 8 WMMA/wave.
template <int CIN>
__global__ void gemm_bn_relu_kernel(const _Float16* __restrict__ X,
                                    const _Float16* __restrict__ Wp,
                                    const float* __restrict__ bnA,
                                    const float* __restrict__ bnC,
                                    _Float16* __restrict__ Y) {
    constexpr int KSTEPS = CIN / 32;
    int lane = threadIdx.x & 31;
    int wave = threadIdx.x >> 5;
    long n0  = ((long)blockIdx.x * 4 + wave) * 16;

    v16h bf[KSTEPS];
#pragma unroll
    for (int ks = 0; ks < KSTEPS; ++ks)
        bf[ks] = load_B_frag(X, CIN, n0, ks * 32, lane);

    int N   = lane & 15;
    int mhi = (lane >> 4) * 8;
    _Float16* dstRow = Y + (n0 + N) * COUT12 + mhi;

#pragma unroll
    for (int mt = 0; mt < COUT12 / 16; ++mt) {
        int m0 = mt * 16;
        v8f acc = {0.f, 0.f, 0.f, 0.f, 0.f, 0.f, 0.f, 0.f};
#pragma unroll
        for (int ks = 0; ks < KSTEPS; ++ks) {
            v16h a = load_A_frag(Wp, CIN, m0, ks * 32, lane);
            acc = __builtin_amdgcn_wmma_f32_16x16x32_f16(false, a, false, bf[ks],
                                                         (short)0, acc, false, false);
        }
        // D tile: lane holds (M = mhi + r, N) in acc[r] -> 8 contiguous channels
#pragma unroll
        for (int r = 0; r < 8; ++r) {
            int ch  = m0 + mhi + r;
            float y = acc[r] * bnA[ch] + bnC[ch];
            dstRow[m0 + r] = (_Float16)(y > 0.f ? y : 0.f);
        }
    }
}

// ---------------- 6) last conv + bias + max over K (fused), B-frag reuse ----------------
// Wave owns one s (all K=32 columns) and 64 of the 128 out-channels:
// 4 B fragments loaded once, 4 channel tiles x 2 col-halves x 2 k-steps = 16 WMMA.
__global__ void conv3_max_kernel(const _Float16* __restrict__ Y2,
                                 const _Float16* __restrict__ Wlp,
                                 const float* __restrict__ bl,
                                 float* __restrict__ out_np) {
    int lane = threadIdx.x & 31;
    int wave = threadIdx.x >> 5;
    int chg  = blockIdx.x;                   // 0..1 : channel halves (64 each)
    int s    = blockIdx.y * 4 + wave;
    int b    = blockIdx.z;
    long col = ((long)b * SS + s) * KK;      // 32 columns = the K group

    v16h bf[2][2];                           // [col-half][k-step]
#pragma unroll
    for (int h = 0; h < 2; ++h)
#pragma unroll
        for (int ks = 0; ks < 2; ++ks)
            bf[h][ks] = load_B_frag(Y2, CIN2P, col + h * 16, ks * 32, lane);

#pragma unroll
    for (int mt = 0; mt < 4; ++mt) {
        int m0 = chg * 64 + mt * 16;
        v8f acc0 = {0.f, 0.f, 0.f, 0.f, 0.f, 0.f, 0.f, 0.f};
        v8f acc1 = {0.f, 0.f, 0.f, 0.f, 0.f, 0.f, 0.f, 0.f};
#pragma unroll
        for (int ks = 0; ks < 2; ++ks) {
            v16h a = load_A_frag(Wlp, CIN2P, m0, ks * 32, lane);
            acc0 = __builtin_amdgcn_wmma_f32_16x16x32_f16(false, a, false, bf[0][ks],
                                                          (short)0, acc0, false, false);
            acc1 = __builtin_amdgcn_wmma_f32_16x16x32_f16(false, a, false, bf[1][ks],
                                                          (short)0, acc1, false, false);
        }
        // max over the two k-half tiles, then butterfly-max across the 16 N-lanes
        float m[8];
#pragma unroll
        for (int r = 0; r < 8; ++r) m[r] = fmaxf(acc0[r], acc1[r]);
#pragma unroll
        for (int off = 8; off >= 1; off >>= 1) {
#pragma unroll
            for (int r = 0; r < 8; ++r) m[r] = fmaxf(m[r], __shfl_xor(m[r], off, 32));
        }
        if ((lane & 15) == 0) {              // lanes 0 (M=0..7) and 16 (M=8..15)
            int ch0 = m0 + (lane >> 4) * 8;
#pragma unroll
            for (int r = 0; r < 8; ++r)
                out_np[((long)b * COUTL + ch0 + r) * SS + s] = m[r] + bl[ch0 + r];
        }
    }
}

// ---------------- launch ----------------
static inline size_t align_up(size_t x, size_t a) { return (x + a - 1) & ~(a - 1); }

extern "C" void kernel_launch(void* const* d_in, const int* in_sizes, int n_in,
                              void* d_out, int out_size, void* d_ws, size_t ws_size,
                              hipStream_t stream) {
    const float* xyz    = (const float*)d_in[0];
    const float* points = (const float*)d_in[1];
    const float* W1  = (const float*)d_in[2];
    const float* b1  = (const float*)d_in[3];
    const float* g1  = (const float*)d_in[4];
    const float* be1 = (const float*)d_in[5];
    const float* m1  = (const float*)d_in[6];
    const float* v1  = (const float*)d_in[7];
    const float* W2  = (const float*)d_in[8];
    const float* b2  = (const float*)d_in[9];
    const float* g2  = (const float*)d_in[10];
    const float* be2 = (const float*)d_in[11];
    const float* m2  = (const float*)d_in[12];
    const float* v2  = (const float*)d_in[13];
    const float* Wl  = (const float*)d_in[14];
    const float* bl  = (const float*)d_in[15];
    float* out = (float*)d_out;

    // ---- workspace carve-out ----
    char* ws = (char*)d_ws;
    size_t off = 0;
    int*   fidx    = (int*)(ws + off);        off = align_up(off + sizeof(int) * BB * SS, 256);
    float* sampled = (float*)(ws + off);      off = align_up(off + sizeof(float) * BB * SS * 3, 256);
    int*   bidx    = (int*)(ws + off);        off = align_up(off + sizeof(int) * NCOL, 256);
    float* bn1a    = (float*)(ws + off);      off = align_up(off + sizeof(float) * COUT12, 256);
    float* bn1c    = (float*)(ws + off);      off = align_up(off + sizeof(float) * COUT12, 256);
    float* bn2a    = (float*)(ws + off);      off = align_up(off + sizeof(float) * COUT12, 256);
    float* bn2c    = (float*)(ws + off);      off = align_up(off + sizeof(float) * COUT12, 256);
    _Float16* W1p  = (_Float16*)(ws + off);   off = align_up(off + sizeof(_Float16) * COUT12 * CIN1P, 256);
    _Float16* W2p  = (_Float16*)(ws + off);   off = align_up(off + sizeof(_Float16) * COUT12 * CIN2P, 256);
    _Float16* Wlp  = (_Float16*)(ws + off);   off = align_up(off + sizeof(_Float16) * COUTL  * CIN2P, 256);
    _Float16* Xg   = (_Float16*)(ws + off);   off = align_up(off + sizeof(_Float16) * NCOL * CIN1P, 256);
    _Float16* Y1   = (_Float16*)(ws + off);   off = align_up(off + sizeof(_Float16) * NCOL * COUT12, 256);
    _Float16* Y2   = (_Float16*)(ws + off);   off = align_up(off + sizeof(_Float16) * NCOL * COUT12, 256);
    (void)ws_size; (void)in_sizes; (void)n_in; (void)out_size;

    float* out_sxyz = out;                     // (B,3,S)
    float* out_np   = out + (long)BB * 3 * SS; // (B,128,S)

    // 0) pack weights / fold BN
    prep_kernel<<<1, 256, 0, stream>>>(W1, b1, g1, be1, m1, v1,
                                       W2, b2, g2, be2, m2, v2, Wl,
                                       W1p, W2p, Wlp, bn1a, bn1c, bn2a, bn2c);

    // 1) FPS: one block per batch; LDS = 4 arrays of N floats + 8-entry handoff
    size_t fps_smem = sizeof(float) * NN * 4 + 8 * (sizeof(float) + sizeof(int));
    fps_kernel<<<BB, 256, fps_smem, stream>>>(xyz, fidx, sampled);

    // 2) sampled_xyz output
    {
        long tot = (long)BB * 3 * SS;
        sampled_out_kernel<<<(int)((tot + 255) / 256), 256, 0, stream>>>(sampled, out_sxyz);
    }

    // 3) ball query
    ball_kernel<<<(BB * SS + 255) / 256, 256, 0, stream>>>(xyz, sampled, bidx);

    // 4) gather + group
    gather_kernel<<<(int)((NCOL + 255) / 256), 256, 0, stream>>>(xyz, points, sampled, bidx, Xg);

    // 5) conv1: 96 -> 64, BN+ReLU   (wave per n-tile, all 64 channels, 12 WMMA)
    gemm_bn_relu_kernel<CIN1P><<<(unsigned)(NCOL / 64), 128, 0, stream>>>(Xg, W1p, bn1a, bn1c, Y1);

    // 6) conv2: 64 -> 64, BN+ReLU   (8 WMMA per wave)
    gemm_bn_relu_kernel<CIN2P><<<(unsigned)(NCOL / 64), 128, 0, stream>>>(Y1, W2p, bn2a, bn2c, Y2);

    // 7) conv3 (64 -> 128) + bias + max over K, fused (16 WMMA per wave)
    {
        dim3 grid(2, SS / 4, BB);
        conv3_max_kernel<<<grid, 128, 0, stream>>>(Y2, Wlp, bl, out_np);
    }
}
